// MultiHeadAttention_42460046688527
// MI455X (gfx1250) — compile-verified, matrix operand order checked by reading, not run on hardware
//
#include <hip/hip_runtime.h>
#include <hip/hip_bf16.h>

// ---------------------------------------------------------------------------
// MHA block for MI455X (gfx1250): bf16 WMMA GEMMs, LDS-resident score strip,
// TDM (tensor_load_to_lds / tensor_store_from_lds) double-buffered staging.
// B=2, S=2048, D=1024, H=16, DK=64.
// ---------------------------------------------------------------------------

typedef __attribute__((ext_vector_type(16))) __bf16 v16bf;
typedef __attribute__((ext_vector_type(8)))  float  v8f;
typedef __attribute__((ext_vector_type(4)))  unsigned int u32x4;
typedef __attribute__((ext_vector_type(4)))  int i32x4;
typedef __attribute__((ext_vector_type(8)))  int i32x8;

#define BATCH 2
#define SEQ   2048
#define DIM   1024
#define HEADS 16
#define DK    64
#define ROWS  (BATCH * SEQ)   // 4096 GEMM rows

// ---------------------------------------------------------------------------
// TDM support (guarded: falls back to manual staging if builtins absent)
// ---------------------------------------------------------------------------
#if __has_builtin(__builtin_amdgcn_tensor_load_to_lds) && \
    __has_builtin(__builtin_amdgcn_tensor_store_from_lds)
#define HAVE_TDM 1
#endif

typedef const char __attribute__((address_space(3))) lds_cchar;

__device__ __forceinline__ unsigned lds_addr_of(const void* p) {
  return (unsigned)(unsigned long long)(lds_cchar*)p;
}

__device__ __forceinline__ void wait_tensorcnt0() {
#if __has_builtin(__builtin_amdgcn_s_wait_tensorcnt)
  __builtin_amdgcn_s_wait_tensorcnt(0);
#else
  asm volatile("s_wait_tensorcnt 0x0" ::: "memory");
#endif
}
__device__ __forceinline__ void wait_tensorcnt1() {
#if __has_builtin(__builtin_amdgcn_s_wait_tensorcnt)
  __builtin_amdgcn_s_wait_tensorcnt(1);
#else
  asm volatile("s_wait_tensorcnt 0x1" ::: "memory");
#endif
}

#ifdef HAVE_TDM
// D# group 0: count=1, lds_addr, 57-bit global addr, type=2 ("image")
__device__ __forceinline__ u32x4 tdm_group0(unsigned lds_addr,
                                            unsigned long long gaddr) {
  u32x4 g;
  g.x = 1u;
  g.y = lds_addr;
  g.z = (unsigned)gaddr;
  g.w = (unsigned)((gaddr >> 32) & 0x1FFFFFFull) | (2u << 30);
  return g;
}
// D# group 1: 2D tile (cols x rows), data_size code 0/1/2/3 = 1/2/4/8 B
__device__ __forceinline__ i32x8 tdm_group1(unsigned ds_code, unsigned cols,
                                            unsigned rows, unsigned stride,
                                            unsigned tile_c, unsigned tile_r) {
  i32x8 g;
  g[0] = (int)(ds_code << 16);                               // mask=0, flags=0
  g[1] = (int)((cols & 0xFFFFu) << 16);                      // tensor_dim0 lo
  g[2] = (int)((cols >> 16) | ((rows & 0xFFFFu) << 16));     // d0 hi | d1 lo
  g[3] = (int)((rows >> 16) | ((tile_c & 0xFFFFu) << 16));   // d1 hi | tile_d0
  g[4] = (int)(tile_r & 0xFFFFu);                            // tile_d1, d2=0
  g[5] = (int)stride;                                        // dim0_stride lo
  g[6] = 0;                                                  // stride hi, d1str
  g[7] = 0;
  return g;
}

__device__ __forceinline__ void tdm_load_2d(void* lds_dst, const void* gsrc,
                                            unsigned ds_code, unsigned cols,
                                            unsigned rows, unsigned stride) {
  u32x4 g0 = tdm_group0(lds_addr_of(lds_dst), (unsigned long long)(size_t)gsrc);
  i32x8 g1 = tdm_group1(ds_code, cols, rows, stride, cols, rows);
  i32x4 z4 = {};
#if __has_include(<hip/amd_detail/amd_gfx1250_TDM.h>)
  i32x8 z8 = {};
  __builtin_amdgcn_tensor_load_to_lds(g0, g1, z4, z4, z8, 0);
#else
  __builtin_amdgcn_tensor_load_to_lds(g0, g1, z4, z4, 0);
#endif
}

__device__ __forceinline__ void tdm_store_2d(const void* lds_src, void* gdst,
                                             unsigned ds_code, unsigned cols,
                                             unsigned rows, unsigned stride) {
  u32x4 g0 = tdm_group0(lds_addr_of(lds_src), (unsigned long long)(size_t)gdst);
  i32x8 g1 = tdm_group1(ds_code, cols, rows, stride, cols, rows);
  i32x4 z4 = {};
#if __has_include(<hip/amd_detail/amd_gfx1250_TDM.h>)
  i32x8 z8 = {};
  __builtin_amdgcn_tensor_store_from_lds(g0, g1, z4, z4, z8, 0);
#else
  __builtin_amdgcn_tensor_store_from_lds(g0, g1, z4, z4, 0);
#endif
}
#endif  // HAVE_TDM

// ---- WMMA fragment helpers (wave32, 16x16x32 bf16) -------------------------
// A 16x32 (bf16): lane 0-15 row=lane, K = {0..7,16..23}; lane 16-31 row=lane-16,
// K = {8..15,24..31}.  Element e -> K = (e<8 ? e : e+8) + (hi ? 8 : 0).
__device__ __forceinline__ int kmap(int e, int hi) {
  return (e < 8 ? e : e + 8) + (hi ? 8 : 0);
}

__device__ __forceinline__ v16bf frag_row_bf(const __bf16* t, int ld, int row,
                                             int k0, int hi) {
  v16bf f;
#pragma unroll
  for (int e = 0; e < 16; ++e) f[e] = t[row * ld + k0 + kmap(e, hi)];
  return f;
}

__device__ __forceinline__ v16bf frag_col_bf(const __bf16* t, int ld, int col,
                                             int k0, int hi) {
  v16bf f;
#pragma unroll
  for (int e = 0; e < 16; ++e) f[e] = t[(k0 + kmap(e, hi)) * ld + col];
  return f;
}

__device__ __forceinline__ v16bf frag_row_f32(const float* t, int ld, int row,
                                              int k0, int hi) {
  v16bf f;
#pragma unroll
  for (int e = 0; e < 16; ++e) f[e] = (__bf16)t[row * ld + k0 + kmap(e, hi)];
  return f;
}

__device__ __forceinline__ v8f wmma_bf16(v16bf a, v16bf b, v8f c) {
  return __builtin_amdgcn_wmma_f32_16x16x32_bf16(
      false, a, false, b, (short)0, c, false, false);
}

// ---------------------------------------------------------------------------
// Kernel 1: projection GEMM  dst[b,h,s,dk] = X[m,:] . W[n,:] + bias[n]
// ---------------------------------------------------------------------------
__global__ void proj_kernel(const float* __restrict__ X,
                            const float* __restrict__ W,
                            const float* __restrict__ bias,
                            __bf16* __restrict__ dst) {
  __shared__ __bf16 As[64][32];
  __shared__ __bf16 Bs[64][32];
  const int tid  = threadIdx.x;
  const int wave = tid >> 5;
  const int lane = tid & 31;
  const int l16  = lane & 15;
  const int hi   = lane >> 4;
  const int n0 = blockIdx.x * 64;
  const int m0 = blockIdx.y * 64;

  v8f acc[4] = {};

  for (int k0 = 0; k0 < DIM; k0 += 32) {
    __syncthreads();
    for (int i = tid; i < 64 * 32; i += 128) {
      int r = i >> 5, c = i & 31;
      As[r][c] = (__bf16)X[(size_t)(m0 + r) * DIM + k0 + c];
      Bs[r][c] = (__bf16)W[(size_t)(n0 + r) * DIM + k0 + c];
    }
    if (k0 + 32 < DIM && tid < 64) {   // gfx1250 global_prefetch of next K tile
      __builtin_prefetch(&X[(size_t)(m0 + tid) * DIM + k0 + 32], 0, 1);
      __builtin_prefetch(&W[(size_t)(n0 + tid) * DIM + k0 + 32], 0, 1);
    }
    __syncthreads();

    v16bf a = frag_row_bf(&As[0][0], 32, wave * 16 + l16, 0, hi);
#pragma unroll
    for (int ns = 0; ns < 4; ++ns) {
      v16bf b = frag_row_bf(&Bs[0][0], 32, ns * 16 + l16, 0, hi);
      acc[ns] = wmma_bf16(a, b, acc[ns]);
    }
  }

#pragma unroll
  for (int ns = 0; ns < 4; ++ns) {
    int n  = n0 + ns * 16 + l16;
    float bv = bias[n];
    int h  = n >> 6, dk = n & 63;
#pragma unroll
    for (int r = 0; r < 8; ++r) {
      int m = m0 + wave * 16 + r + hi * 8;
      int b = m >> 11, s = m & 2047;
      dst[(((size_t)(b * HEADS + h) * SEQ + s) * DK) + dk] =
          (__bf16)(acc[ns][r] + bv);
    }
  }
}

// ---------------------------------------------------------------------------
// Kernel 2: attention. grid (S/32, B*H), 128 threads.
// LDS: 32x2048 f32 score strip (256KB) + q tile + double-buffered k/v tiles.
// ---------------------------------------------------------------------------
#define SM_SCORES (32 * SEQ * 4)          // 262144 B
#define SM_Q      (32 * DK * 2)           //   4096 B
#define SM_KV     (64 * DK * 2)           //   8192 B (x2 buffers)
#define ATTN_SMEM (SM_SCORES + SM_Q + 2 * SM_KV + 512)

__global__ void attn_kernel(const __bf16* __restrict__ qbuf,
                            const __bf16* __restrict__ kbuf,
                            const __bf16* __restrict__ vbuf,
                            float* __restrict__ attn_out,
                            __bf16* __restrict__ ctx) {
  extern __shared__ char smem[];
  float*  Ssc = (float*)smem;                                // [32][2048]
  __bf16* Qs  = (__bf16*)(smem + SM_SCORES);                 // [32][64]
  __bf16* KV0 = Qs + 32 * DK;                                // [64][64]
  __bf16* KV1 = KV0 + 64 * DK;                               // [64][64]
  float*  red = (float*)(smem + SM_SCORES + SM_Q + 2 * SM_KV);

  const int tid  = threadIdx.x;
  const int wave = tid >> 5;
  const int lane = tid & 31;
  const int l16  = lane & 15;
  const int hi   = lane >> 4;
  const int msub  = wave & 1;
  const int npair = wave >> 1;

  const int bh = blockIdx.y;
  const int q0 = blockIdx.x * 32;
  const __bf16* qb = qbuf + (size_t)bh * SEQ * DK;
  const __bf16* kb = kbuf + (size_t)bh * SEQ * DK;
  const __bf16* vb = vbuf + (size_t)bh * SEQ * DK;
  float* attn_dst = attn_out + ((size_t)bh * SEQ + q0) * SEQ;

  // init scores to masked value
  for (int i = tid; i < 32 * SEQ; i += 128) Ssc[i] = -1e9f;

  // stage q tile (32x64 bf16, contiguous rows)
#ifdef HAVE_TDM
  if (wave == 0) tdm_load_2d(Qs, qb + (size_t)q0 * DK, 1, DK, 32, DK);
#else
  for (int i = tid; i < 32 * DK; i += 128)
    Qs[i] = qb[(size_t)q0 * DK + i];
#endif

  const float scale = 0.125f;        // 1/sqrt(64)
  const int ktmax = (q0 + 31) >> 6;  // last causal key tile (inclusive)

  // ---- Phase 1: scores = q . k^T (double-buffered K tiles) ----
#ifdef HAVE_TDM
  if (wave == 0) tdm_load_2d(KV0, kb, 1, DK, 64, DK);
#else
  for (int i = tid; i < 64 * DK; i += 128) KV0[i] = kb[i];
#endif
  for (int kt = 0; kt <= ktmax; ++kt) {
    const __bf16* KV = (kt & 1) ? KV1 : KV0;
    if (kt + 1 <= ktmax) {
      __bf16* nxt = ((kt + 1) & 1) ? KV1 : KV0;
      const __bf16* src = kb + (size_t)(kt + 1) * 64 * DK;
#ifdef HAVE_TDM
      if (wave == 0) { tdm_load_2d(nxt, src, 1, DK, 64, DK); wait_tensorcnt1(); }
#else
      for (int i = tid; i < 64 * DK; i += 128) nxt[i] = src[i];
#endif
    } else {
#ifdef HAVE_TDM
      if (wave == 0) wait_tensorcnt0();
#endif
    }
    __syncthreads();

    v8f acc[2] = {};
#pragma unroll
    for (int kk = 0; kk < 2; ++kk) {
      v16bf a = frag_row_bf(Qs, DK, msub * 16 + l16, kk * 32, hi);
#pragma unroll
      for (int j = 0; j < 2; ++j) {
        int nsub = npair * 2 + j;
        v16bf b = frag_row_bf(KV, DK, nsub * 16 + l16, kk * 32, hi);
        acc[j] = wmma_bf16(a, b, acc[j]);
      }
    }
#pragma unroll
    for (int j = 0; j < 2; ++j) {
      int nsub = npair * 2 + j;
      int col  = kt * 64 + nsub * 16 + l16;
#pragma unroll
      for (int r = 0; r < 8; ++r) {
        int row = msub * 16 + r + hi * 8;
        Ssc[row * SEQ + col] = (col <= q0 + row) ? acc[j][r] * scale : -1e9f;
      }
    }
    __syncthreads();
  }

  // ---- Phase 2: softmax over LDS strip ----
  {
    const int row = tid & 31;
    const int qtr = tid >> 5;
    float mx = -1e30f;
    for (int c = qtr; c < SEQ; c += 4) mx = fmaxf(mx, Ssc[row * SEQ + c]);
    red[row * 4 + qtr] = mx;
    __syncthreads();
    float rm = fmaxf(fmaxf(red[row * 4 + 0], red[row * 4 + 1]),
                     fmaxf(red[row * 4 + 2], red[row * 4 + 3]));
    __syncthreads();
    float sm = 0.f;
    for (int c = qtr; c < SEQ; c += 4) {
      float e = __expf(Ssc[row * SEQ + c] - rm);
      Ssc[row * SEQ + c] = e;
      sm += e;
    }
    red[row * 4 + qtr] = sm;
    __syncthreads();
    float inv = 1.0f / (red[row * 4 + 0] + red[row * 4 + 1] +
                        red[row * 4 + 2] + red[row * 4 + 3]);
    for (int c = qtr; c < SEQ; c += 4) Ssc[row * SEQ + c] *= inv;
  }
  __syncthreads();

  // ---- Phase 3: ctx = attn . V (double-buffered V tiles; reads Ssc only) ----
  v8f cacc[2] = {};
#ifdef HAVE_TDM
  if (wave == 0) tdm_load_2d(KV0, vb, 1, DK, 64, DK);
#else
  for (int i = tid; i < 64 * DK; i += 128) KV0[i] = vb[i];
#endif
  for (int kt = 0; kt <= ktmax; ++kt) {
    const __bf16* KV = (kt & 1) ? KV1 : KV0;
    if (kt + 1 <= ktmax) {
      __bf16* nxt = ((kt + 1) & 1) ? KV1 : KV0;
      const __bf16* src = vb + (size_t)(kt + 1) * 64 * DK;
#ifdef HAVE_TDM
      if (wave == 0) { tdm_load_2d(nxt, src, 1, DK, 64, DK); wait_tensorcnt1(); }
#else
      for (int i = tid; i < 64 * DK; i += 128) nxt[i] = src[i];
#endif
    } else {
#ifdef HAVE_TDM
      if (wave == 0) wait_tensorcnt0();
#endif
    }
    __syncthreads();
#pragma unroll
    for (int kk = 0; kk < 2; ++kk) {
      v16bf a = frag_row_f32(Ssc, SEQ, msub * 16 + l16, kt * 64 + kk * 32, hi);
#pragma unroll
      for (int j = 0; j < 2; ++j) {
        int nsub = npair * 2 + j;
        v16bf b = frag_col_bf(KV, DK, nsub * 16 + l16, kk * 32, hi);
        cacc[j] = wmma_bf16(a, b, cacc[j]);
      }
    }
    __syncthreads();
  }

  // store ctx as bf16 in [B,S,D] layout for the out-proj GEMM
  {
    const int h = bh & (HEADS - 1), bb = bh >> 4;
#pragma unroll
    for (int j = 0; j < 2; ++j) {
      int dk = npair * 32 + j * 16 + l16;
#pragma unroll
      for (int r = 0; r < 8; ++r) {
        int s = q0 + msub * 16 + r + hi * 8;
        ctx[((size_t)bb * SEQ + s) * DIM + h * DK + dk] = (__bf16)cacc[j][r];
      }
    }
  }

  // ---- stream the 32x2048 f32 probability strip to global ----
#ifdef HAVE_TDM
  if (wave == 0) tdm_store_2d(Ssc, attn_dst, 2, SEQ, 32, SEQ);
  // S_ENDPGM performs an implicit wait-idle (drains TENSORcnt)
#else
  for (int i = tid; i < 32 * SEQ; i += 128) attn_dst[i] = Ssc[i];
#endif
}

// ---------------------------------------------------------------------------
// Kernel 3: output projection  tmp[m,n] = ctx[m,:] . Wo[n,:] + bo[n]   (f32)
// ---------------------------------------------------------------------------
__global__ void outproj_kernel(const __bf16* __restrict__ X,
                               const float* __restrict__ W,
                               const float* __restrict__ bias,
                               float* __restrict__ tmp) {
  __shared__ __bf16 As[64][32];
  __shared__ __bf16 Bs[64][32];
  const int tid  = threadIdx.x;
  const int wave = tid >> 5;
  const int lane = tid & 31;
  const int l16  = lane & 15;
  const int hi   = lane >> 4;
  const int n0 = blockIdx.x * 64;
  const int m0 = blockIdx.y * 64;

  v8f acc[4] = {};
  for (int k0 = 0; k0 < DIM; k0 += 32) {
    __syncthreads();
    for (int i = tid; i < 64 * 32; i += 128) {
      int r = i >> 5, c = i & 31;
      As[r][c] = X[(size_t)(m0 + r) * DIM + k0 + c];
      Bs[r][c] = (__bf16)W[(size_t)(n0 + r) * DIM + k0 + c];
    }
    if (k0 + 32 < DIM && tid < 64) {
      __builtin_prefetch(&X[(size_t)(m0 + tid) * DIM + k0 + 32], 0, 1);
      __builtin_prefetch(&W[(size_t)(n0 + tid) * DIM + k0 + 32], 0, 1);
    }
    __syncthreads();

    v16bf a = frag_row_bf(&As[0][0], 32, wave * 16 + l16, 0, hi);
#pragma unroll
    for (int ns = 0; ns < 4; ++ns) {
      v16bf b = frag_row_bf(&Bs[0][0], 32, ns * 16 + l16, 0, hi);
      acc[ns] = wmma_bf16(a, b, acc[ns]);
    }
  }

#pragma unroll
  for (int ns = 0; ns < 4; ++ns) {
    int n = n0 + ns * 16 + l16;
    float bv = bias[n];
#pragma unroll
    for (int r = 0; r < 8; ++r) {
      int m = m0 + wave * 16 + r + hi * 8;
      tmp[(size_t)m * DIM + n] = acc[ns][r] + bv;
    }
  }
}

// ---------------------------------------------------------------------------
// Kernel 4: residual + LayerNorm. One 256-thread block per row (1024 cols).
// ---------------------------------------------------------------------------
__global__ void ln_kernel(const float* __restrict__ tmp,
                          const float* __restrict__ resid,
                          const float* __restrict__ gamma,
                          const float* __restrict__ beta,
                          float* __restrict__ out) {
  __shared__ float sdata[256];
  const int tid = threadIdx.x;
  const size_t rowbase = (size_t)blockIdx.x * DIM;

  float x[4];
  float s = 0.f;
#pragma unroll
  for (int i = 0; i < 4; ++i) {
    int c = tid + i * 256;
    x[i] = tmp[rowbase + c] + resid[rowbase + c];
    s += x[i];
  }
  sdata[tid] = s;
  __syncthreads();
  for (int off = 128; off > 0; off >>= 1) {
    if (tid < off) sdata[tid] += sdata[tid + off];
    __syncthreads();
  }
  const float mu = sdata[0] * (1.0f / (float)DIM);
  __syncthreads();

  float v = 0.f;
#pragma unroll
  for (int i = 0; i < 4; ++i) {
    float d = x[i] - mu;
    v += d * d;
  }
  sdata[tid] = v;
  __syncthreads();
  for (int off = 128; off > 0; off >>= 1) {
    if (tid < off) sdata[tid] += sdata[tid + off];
    __syncthreads();
  }
  const float inv = rsqrtf(sdata[0] * (1.0f / (float)DIM) + 1e-5f);

#pragma unroll
  for (int i = 0; i < 4; ++i) {
    int c = tid + i * 256;
    out[rowbase + c] = (x[i] - mu) * inv * gamma[c] + beta[c];
  }
}

// ---------------------------------------------------------------------------
// Launcher
// ---------------------------------------------------------------------------
extern "C" void kernel_launch(void* const* d_in, const int* in_sizes, int n_in,
                              void* d_out, int out_size, void* d_ws,
                              size_t ws_size, hipStream_t stream) {
  (void)in_sizes; (void)n_in; (void)out_size; (void)ws_size;

  const float* Q  = (const float*)d_in[0];
  const float* K  = (const float*)d_in[1];
  const float* V  = (const float*)d_in[2];
  // d_in[3] = attn_mask (causal; computed analytically in-kernel)
  const float* Wq = (const float*)d_in[4];
  const float* bq = (const float*)d_in[5];
  const float* Wk = (const float*)d_in[6];
  const float* bk = (const float*)d_in[7];
  const float* Wv = (const float*)d_in[8];
  const float* bv = (const float*)d_in[9];
  const float* Wo = (const float*)d_in[10];
  const float* bo = (const float*)d_in[11];
  const float* gamma = (const float*)d_in[12];
  const float* beta  = (const float*)d_in[13];

  float* out_ln   = (float*)d_out;                          // [B,S,D]
  float* out_attn = out_ln + (size_t)BATCH * SEQ * DIM;     // [B,H,S,S]

  char* ws = (char*)d_ws;
  const size_t NE = (size_t)BATCH * SEQ * DIM;              // 4194304 elems
  __bf16* qb   = (__bf16*)(ws);
  __bf16* kb   = (__bf16*)(ws + 2 * NE);
  __bf16* vb   = (__bf16*)(ws + 4 * NE);
  __bf16* ctxb = (__bf16*)(ws + 6 * NE);
  float*  tmp  = (float*)(ws + 8 * NE);

  dim3 gemm_grid(DIM / 64, ROWS / 64);                      // (16, 64)
  proj_kernel<<<gemm_grid, 128, 0, stream>>>(Q, Wq, bq, qb);
  proj_kernel<<<gemm_grid, 128, 0, stream>>>(K, Wk, bk, kb);
  proj_kernel<<<gemm_grid, 128, 0, stream>>>(V, Wv, bv, vb);

  dim3 attn_grid(SEQ / 32, BATCH * HEADS);                  // (64, 32)
  attn_kernel<<<attn_grid, 128, ATTN_SMEM, stream>>>(qb, kb, vb, out_attn,
                                                     ctxb);

  outproj_kernel<<<gemm_grid, 128, 0, stream>>>(ctxb, Wo, bo, tmp);

  ln_kernel<<<ROWS, 256, 0, stream>>>(tmp, Q, gamma, beta, out_ln);
}
